// BothMamba_91036126806278
// MI455X (gfx1250) — matrix operand order repaired and checked by reading, not compiled, not run on hardware
//
#include <hip/hip_runtime.h>
#include <hip/hip_bf16.h>

typedef __attribute__((ext_vector_type(16))) _Float16 v16h;
typedef __attribute__((ext_vector_type(8)))  _Float16 v8h;
typedef __attribute__((ext_vector_type(8)))  float    v8f;

#define DEV __device__ __forceinline__

// ---------------- problem constants ----------------
#define B_    8
#define C_    96
#define HW_   4096          // H*W = 64*64, also L for spatial mamba
#define DIN_  192           // spatial d_in = 2*C
#define NST_  16            // D_STATE
#define DTR_  6             // spatial dt_rank
#define XPN_  48            // padded (dt_rank + 2*N) = 38 -> 48
#define TOK_  8             // spectral sequence length
#define GCN_  12            // spectral d_model
#define SDIN_ 24            // spectral d_in
#define CHL_  256           // scan chunk length
#define NCH_  16            // number of chunks (4096/256)

DEV float sigmoid_(float x){ return 1.f/(1.f + __expf(-x)); }
DEV float silu_(float x){ return x * sigmoid_(x); }
DEV float softplus_(float x){ return (x > 20.f) ? x : __logf(1.f + __expf(x)); }

// ---------------------------------------------------------------------------
// Wave-level 16x16 output-tile GEMM accumulator over K (f16 inputs, f32 acc).
// A: row-major [16][LDA] f16.  Bt: W^T row-major [N][LDB] f16.
// Fragment layouts follow CDNA5 16-bit WMMA VGPR striping:
//   A lane(m=lane&15,g=lane>>4): halves K = k+8g..k+8g+7 and k+16+8g..+7
//   B lane(n=lane&15,g=lane>>4): halves K = k+8g..k+8g+7 and k+16+8g..+7 (from W^T row n)
// ---------------------------------------------------------------------------
template<int K, int LDA, int LDB>
DEV v8f wmma_tile(const _Float16* A, const _Float16* Bt, int nbase, int lane, v8f c) {
  const int r = lane & 15, g = lane >> 4;
  #pragma unroll
  for (int k = 0; k < K; k += 32) {
    const _Float16* pa = A + r*LDA + k + 8*g;
    v8h a0 = *(const v8h*)pa;
    v8h a1 = *(const v8h*)(pa + 16);
    v16h af = __builtin_shufflevector(a0, a1, 0,1,2,3,4,5,6,7,8,9,10,11,12,13,14,15);
    const _Float16* pb = Bt + (size_t)(nbase + r)*LDB + k + 8*g;
    v8h b0 = *(const v8h*)pb;
    v8h b1 = *(const v8h*)(pb + 16);
    v16h bf = __builtin_shufflevector(b0, b1, 0,1,2,3,4,5,6,7,8,9,10,11,12,13,14,15);
    c = __builtin_amdgcn_wmma_f32_16x16x32_f16(false, af, false, bf, (short)0, c,
                                               false, false);
  }
  return c;
}

// ---------------------------------------------------------------------------
// Pack f32 weights -> transposed f16 ([N][K]) fragment-friendly layout.
// ---------------------------------------------------------------------------
__global__ void pack_weights_kernel(const float* __restrict__ W1,   // (96,192)
                                    const float* __restrict__ W2,   // (192,96)
                                    const float* __restrict__ Win,  // (96,384)
                                    const float* __restrict__ Wx,   // (192,38)
                                    const float* __restrict__ Wout, // (192,96)
                                    _Float16* __restrict__ W1T,     // [192][96]
                                    _Float16* __restrict__ W2T,     // [96][192]
                                    _Float16* __restrict__ WinT,    // [384][96]
                                    _Float16* __restrict__ WxT,     // [48][192]
                                    _Float16* __restrict__ WoutT) { // [96][192]
  int i = blockIdx.x*blockDim.x + threadIdx.x;
  if (i < 96*192)  { int k = i/192, n = i%192; W1T [n*96  + k] = (_Float16)W1[i]; }
  if (i < 192*96)  { int k = i/96,  n = i%96;  W2T [n*192 + k] = (_Float16)W2[i]; }
  if (i < 96*384)  { int k = i/384, n = i%384; WinT[n*96  + k] = (_Float16)Win[i]; }
  if (i < 48*192)  { int n = i/192, k = i%192;
                     WxT[n*192 + k] = (n < 38) ? (_Float16)Wx[k*38 + n] : (_Float16)0.f; }
  if (i < 192*96)  { int k = i/96,  n = i%96;  WoutT[n*192 + k] = (_Float16)Wout[i]; }
}

// ---------------------------------------------------------------------------
// Spatial: MLP (relu) + MLP2 + mamba in-projection, all WMMA, fused per
// 16-token tile. 4 waves / block, each wave owns one tile.
//   xf[l][c] = x_spa[b][c][l]
//   hm = relu(xf@W1+b1)@W2+b2 ; xz = hm@Win ; xc -> f32 ws, z -> f16 ws
// ---------------------------------------------------------------------------
__global__ void spa_mlp_inproj_kernel(const float* __restrict__ x_spa,
                                      const _Float16* __restrict__ W1T,
                                      const float* __restrict__ b1,
                                      const _Float16* __restrict__ W2T,
                                      const float* __restrict__ b2,
                                      const _Float16* __restrict__ WinT,
                                      float* __restrict__ xc_out,      // [B*L][192]
                                      _Float16* __restrict__ zh_out) { // [B*L][192]
  __shared__ alignas(16) _Float16 sA [4][16][96];
  __shared__ alignas(16) _Float16 sH1[4][16][192];
  __shared__ alignas(16) _Float16 sH2[4][16][96];
  const int wave = threadIdx.x >> 5, lane = threadIdx.x & 31;
  const int tile = blockIdx.x*4 + wave;           // 0..2047
  const int token0 = tile*16;                     // b*L + l
  const int b = token0 >> 12, l0 = token0 & 4095;

  for (int i = lane; i < 16*96; i += 32) {
    int r = i/96, c = i%96;
    sA[wave][r][c] = (_Float16)x_spa[((size_t)b*C_ + c)*HW_ + (l0 + r)];
  }
  __syncthreads();

  for (int nt = 0; nt < 12; ++nt) {               // GEMM1: K=96, N=192, + relu
    v8f acc = {};
    acc = wmma_tile<96,96,96>(&sA[wave][0][0], W1T, nt*16, lane, acc);
    int n = nt*16 + (lane & 15); float bias = b1[n];
    #pragma unroll
    for (int v = 0; v < 8; ++v) {
      int m = v + 8*(lane >> 4);
      sH1[wave][m][n] = (_Float16)fmaxf(acc[v] + bias, 0.f);
    }
  }
  __syncthreads();

  for (int nt = 0; nt < 6; ++nt) {                // GEMM2: K=192, N=96
    v8f acc = {};
    acc = wmma_tile<192,192,192>(&sH1[wave][0][0], W2T, nt*16, lane, acc);
    int n = nt*16 + (lane & 15); float bias = b2[n];
    #pragma unroll
    for (int v = 0; v < 8; ++v) {
      int m = v + 8*(lane >> 4);
      sH2[wave][m][n] = (_Float16)(acc[v] + bias);
    }
  }
  __syncthreads();

  for (int nt = 0; nt < 24; ++nt) {               // GEMM3: K=96, N=384
    v8f acc = {};
    acc = wmma_tile<96,96,96>(&sH2[wave][0][0], WinT, nt*16, lane, acc);
    int n = nt*16 + (lane & 15);
    #pragma unroll
    for (int v = 0; v < 8; ++v) {
      int m = v + 8*(lane >> 4);
      size_t row = (size_t)token0 + m;
      if (n < DIN_) xc_out[row*DIN_ + n] = acc[v];
      else          zh_out[row*DIN_ + (n - DIN_)] = (_Float16)acc[v];
    }
  }
}

// ---------------------------------------------------------------------------
// Depthwise causal conv (width 4) along L + silu. Emits f32 and packed f16.
// ---------------------------------------------------------------------------
__global__ void spa_conv_kernel(const float* __restrict__ xc,
                                const float* __restrict__ convw,  // (192,4)
                                const float* __restrict__ convb,
                                float* __restrict__ xcs,
                                _Float16* __restrict__ xcs_h) {
  size_t idx = (size_t)blockIdx.x*blockDim.x + threadIdx.x;
  if (idx >= (size_t)B_*HW_*DIN_) return;
  int d = idx % DIN_; size_t bl = idx / DIN_;
  int l = bl & 4095; size_t b = bl >> 12;
  float acc = convb[d];
  #pragma unroll
  for (int j = 0; j < 4; ++j) {
    int ls = l + j - 3;
    float xv = (ls >= 0) ? xc[(b*HW_ + ls)*DIN_ + d] : 0.f;
    acc = fmaf(xv, convw[d*4 + j], acc);
  }
  float s = silu_(acc);
  xcs[idx] = s;
  xcs_h[idx] = (_Float16)s;
}

// ---------------------------------------------------------------------------
// x-projection (WMMA, N padded to 48) + dt-projection + softplus.
// Emits dt (f32), Bm, Cm.
// ---------------------------------------------------------------------------
__global__ void spa_xproj_kernel(const _Float16* __restrict__ xcs_h,
                                 const _Float16* __restrict__ WxT, // [48][192]
                                 const float* __restrict__ Wdt,    // (6,192)
                                 const float* __restrict__ bdt,
                                 float* __restrict__ dt_out,
                                 float* __restrict__ Bm,
                                 float* __restrict__ Cm) {
  __shared__ float sD[4][16][XPN_];
  const int wave = threadIdx.x >> 5, lane = threadIdx.x & 31;
  const int tile = blockIdx.x*4 + wave;
  const int token0 = tile*16;
  const _Float16* A = xcs_h + (size_t)token0*DIN_;

  for (int nt = 0; nt < 3; ++nt) {
    v8f acc = {};
    acc = wmma_tile<192,192,192>(A, WxT, nt*16, lane, acc);
    int n = nt*16 + (lane & 15);
    #pragma unroll
    for (int v = 0; v < 8; ++v) { int m = v + 8*(lane >> 4); sD[wave][m][n] = acc[v]; }
  }
  __syncthreads();

  for (int i = lane; i < 16*16; i += 32) {
    int m = i >> 4, n = i & 15;
    size_t row = (size_t)token0 + m;
    Bm[row*NST_ + n] = sD[wave][m][DTR_ + n];
    Cm[row*NST_ + n] = sD[wave][m][DTR_ + NST_ + n];
  }
  for (int i = lane; i < 16*DIN_; i += 32) {
    int m = i / DIN_, d = i % DIN_;
    float a = bdt[d];
    #pragma unroll
    for (int r = 0; r < DTR_; ++r) a = fmaf(sD[wave][m][r], Wdt[r*DIN_ + d], a);
    dt_out[((size_t)token0 + m)*DIN_ + d] = softplus_(a);
  }
}

// ---------------------------------------------------------------------------
// Chunked selective scan:  h_t = exp(dt_t*A_n) h_{t-1} + dt_t u_t B_t[n]
// Phase 1: per (b,chunk,d) local scan with h0=0, track decay product P.
// Phase 2: per (b,d) carry propagation over 16 chunks (tiny, sequential).
// Phase 3: replay chunks from true h_in, emit y_t = h.C_t + u_t D.
// ---------------------------------------------------------------------------
__global__ void spa_scan_phase1(const float* __restrict__ dt,
                                const float* __restrict__ u,
                                const float* __restrict__ Bm,
                                const float* __restrict__ Alog,
                                float* __restrict__ carryH,
                                float* __restrict__ carryP) {
  const int b = blockIdx.x >> 4, ch = blockIdx.x & 15, d = threadIdx.x;
  float A[NST_], h[NST_], P[NST_];
  #pragma unroll
  for (int n = 0; n < NST_; ++n) { A[n] = -__expf(Alog[d*NST_ + n]); h[n] = 0.f; P[n] = 1.f; }
  const int t0 = ch*CHL_;
  for (int t = t0; t < t0 + CHL_; ++t) {
    size_t base = (size_t)b*HW_ + t;
    float dtv = dt[base*DIN_ + d];
    float du  = dtv * u[base*DIN_ + d];
    const float* Bp = Bm + base*NST_;
    __builtin_prefetch(dt + (base + 4)*DIN_ + d);
    __builtin_prefetch(u  + (base + 4)*DIN_ + d);
    #pragma unroll
    for (int n = 0; n < NST_; ++n) {
      float a = __expf(dtv*A[n]);
      h[n] = fmaf(a, h[n], du*Bp[n]);
      P[n] *= a;
    }
  }
  size_t o = ((size_t)(b*NCH_ + ch)*DIN_ + d)*NST_;
  #pragma unroll
  for (int n = 0; n < NST_; ++n) { carryH[o + n] = h[n]; carryP[o + n] = P[n]; }
}

__global__ void spa_scan_phase2(const float* __restrict__ carryH,
                                const float* __restrict__ carryP,
                                float* __restrict__ hin) {
  const int b = blockIdx.x, d = threadIdx.x;
  float h[NST_];
  #pragma unroll
  for (int n = 0; n < NST_; ++n) h[n] = 0.f;
  for (int ch = 0; ch < NCH_; ++ch) {
    size_t o = ((size_t)(b*NCH_ + ch)*DIN_ + d)*NST_;
    #pragma unroll
    for (int n = 0; n < NST_; ++n) {
      hin[o + n] = h[n];
      h[n] = fmaf(carryP[o + n], h[n], carryH[o + n]);
    }
  }
}

__global__ void spa_scan_phase3(const float* __restrict__ dt,
                                const float* __restrict__ u,
                                const float* __restrict__ Bm,
                                const float* __restrict__ Cm,
                                const float* __restrict__ Alog,
                                const float* __restrict__ Dp,
                                const float* __restrict__ hin,
                                float* __restrict__ ys) {
  const int b = blockIdx.x >> 4, ch = blockIdx.x & 15, d = threadIdx.x;
  float A[NST_], h[NST_];
  #pragma unroll
  for (int n = 0; n < NST_; ++n) A[n] = -__expf(Alog[d*NST_ + n]);
  size_t o = ((size_t)(b*NCH_ + ch)*DIN_ + d)*NST_;
  #pragma unroll
  for (int n = 0; n < NST_; ++n) h[n] = hin[o + n];
  const float Dv = Dp[d];
  const int t0 = ch*CHL_;
  for (int t = t0; t < t0 + CHL_; ++t) {
    size_t base = (size_t)b*HW_ + t;
    float dtv = dt[base*DIN_ + d];
    float uv  = u [base*DIN_ + d];
    float du  = dtv*uv;
    const float* Bp = Bm + base*NST_;
    const float* Cp = Cm + base*NST_;
    __builtin_prefetch(dt + (base + 4)*DIN_ + d);
    __builtin_prefetch(u  + (base + 4)*DIN_ + d);
    float y = 0.f;
    #pragma unroll
    for (int n = 0; n < NST_; ++n) {
      h[n] = fmaf(__expf(dtv*A[n]), h[n], du*Bp[n]);
      y = fmaf(h[n], Cp[n], y);
    }
    ys[base*DIN_ + d] = fmaf(uv, Dv, y);
  }
}

// ---------------------------------------------------------------------------
// Gating (y * silu(z)) then out-projection (WMMA).  m_spa layout [B*HW][C].
// ---------------------------------------------------------------------------
__global__ void spa_out_kernel(const float* __restrict__ ys,
                               const _Float16* __restrict__ zh,
                               const _Float16* __restrict__ WoutT, // [96][192]
                               float* __restrict__ m_out) {
  __shared__ alignas(16) _Float16 sG[4][16][192];
  const int wave = threadIdx.x >> 5, lane = threadIdx.x & 31;
  const int tile = blockIdx.x*4 + wave;
  const int token0 = tile*16;

  for (int i = lane; i < 16*DIN_; i += 32) {
    int m = i / DIN_, d = i % DIN_;
    size_t row = (size_t)token0 + m;
    float yv = ys[row*DIN_ + d];
    float zv = (float)zh[row*DIN_ + d];
    sG[wave][m][d] = (_Float16)(yv * silu_(zv));
  }
  __syncthreads();

  for (int nt = 0; nt < 6; ++nt) {
    v8f acc = {};
    acc = wmma_tile<192,192,192>(&sG[wave][0][0], WoutT, nt*16, lane, acc);
    int n = nt*16 + (lane & 15);
    #pragma unroll
    for (int v = 0; v < 8; ++v) {
      int m = v + 8*(lane >> 4);
      m_out[((size_t)token0 + m)*C_ + n] = acc[v];
    }
  }
}

// ---------------------------------------------------------------------------
// Spectral Mamba: one wave per pixel, streaming over the 8-token sequence.
// Cross-channel reductions via LDS. m2 layout [B*HW][C], c = t*12 + cc.
// ---------------------------------------------------------------------------
__global__ void spe_mamba_kernel(const float* __restrict__ x_spe,
                                 const float* __restrict__ Win,   // (12,48)
                                 const float* __restrict__ convw, // (24,4)
                                 const float* __restrict__ convb,
                                 const float* __restrict__ Wx,    // (24,33)
                                 const float* __restrict__ Wdt,   // (1,24)
                                 const float* __restrict__ bdt,
                                 const float* __restrict__ Alog,  // (24,16)
                                 const float* __restrict__ Dp,
                                 const float* __restrict__ Wout,  // (24,12)
                                 float* __restrict__ m2) {
  __shared__ float sx  [8][GCN_];
  __shared__ float sxcs[8][SDIN_];
  __shared__ float sdbl[8][36];
  __shared__ float sg  [8][SDIN_];
  const int wave = threadIdx.x >> 5, lane = threadIdx.x & 31;
  const size_t p = (size_t)blockIdx.x*8 + wave;   // pixel id, 0..32767
  const int b = (int)(p >> 12), hw = (int)(p & 4095);

  float wix[GCN_], wiz[GCN_], cw[4], A[NST_];
  float cb = 0.f, wdt = 0.f, bd = 0.f, Dd = 0.f;
  if (lane < SDIN_) {
    #pragma unroll
    for (int g = 0; g < GCN_; ++g) { wix[g] = Win[g*48 + lane]; wiz[g] = Win[g*48 + 24 + lane]; }
    #pragma unroll
    for (int j = 0; j < 4; ++j) cw[j] = convw[lane*4 + j];
    cb = convb[lane]; wdt = Wdt[lane]; bd = bdt[lane]; Dd = Dp[lane];
    #pragma unroll
    for (int n = 0; n < NST_; ++n) A[n] = -__expf(Alog[lane*NST_ + n]);
  }
  float h[NST_];
  #pragma unroll
  for (int n = 0; n < NST_; ++n) h[n] = 0.f;
  float x1 = 0.f, x2 = 0.f, x3 = 0.f;   // conv history t-1, t-2, t-3
  float xcs = 0.f, zl = 0.f;

  for (int t = 0; t < TOK_; ++t) {
    if (lane < GCN_)
      sx[wave][lane] = x_spe[((size_t)b*C_ + t*GCN_ + lane)*HW_ + hw];
    __syncthreads();
    if (lane < SDIN_) {
      float xl = 0.f; zl = 0.f;
      #pragma unroll
      for (int g = 0; g < GCN_; ++g) {
        float xv = sx[wave][g];
        xl = fmaf(xv, wix[g], xl); zl = fmaf(xv, wiz[g], zl);
      }
      float acc = cb + cw[0]*x3 + cw[1]*x2 + cw[2]*x1 + cw[3]*xl;
      x3 = x2; x2 = x1; x1 = xl;
      xcs = silu_(acc);
      sxcs[wave][lane] = xcs;
    }
    __syncthreads();
    for (int j = lane; j < 33; j += 32) {
      float a = 0.f;
      #pragma unroll
      for (int k = 0; k < SDIN_; ++k) a = fmaf(sxcs[wave][k], Wx[k*33 + j], a);
      sdbl[wave][j] = a;
    }
    __syncthreads();
    if (lane < SDIN_) {
      float dtv = softplus_(fmaf(sdbl[wave][0], wdt, bd));
      float du  = dtv * xcs;
      float y = 0.f;
      #pragma unroll
      for (int n = 0; n < NST_; ++n) {
        h[n] = fmaf(__expf(dtv*A[n]), h[n], du*sdbl[wave][1 + n]);
        y = fmaf(h[n], sdbl[wave][17 + n], y);
      }
      sg[wave][lane] = (y + xcs*Dd) * silu_(zl);
    }
    __syncthreads();
    if (lane < GCN_) {
      float o = 0.f;
      #pragma unroll
      for (int d2 = 0; d2 < SDIN_; ++d2) o = fmaf(sg[wave][d2], Wout[d2*GCN_ + lane], o);
      m2[((size_t)b*HW_ + hw)*C_ + t*GCN_ + lane] = o;
    }
    __syncthreads();
  }
}

// ---------------------------------------------------------------------------
// GroupNorm statistics: one block per (b, group of 24 channels).
// Input layout [B*HW][C]; stats -> {mean, var} per (b,g).
// ---------------------------------------------------------------------------
__global__ void gn_stats_kernel(const float* __restrict__ m, float* __restrict__ stats) {
  const int bg = blockIdx.x, b = bg >> 2, g = bg & 3;
  float s = 0.f, ss = 0.f;
  for (int i = threadIdx.x; i < 24*HW_; i += blockDim.x) {
    int c = 24*g + (i >> 12), hw = i & 4095;
    float v = m[((size_t)b*HW_ + hw)*C_ + c];
    s += v; ss = fmaf(v, v, ss);
  }
  __shared__ float rs[256], rq[256];
  rs[threadIdx.x] = s; rq[threadIdx.x] = ss;
  __syncthreads();
  for (int st = 128; st > 0; st >>= 1) {
    if (threadIdx.x < st) { rs[threadIdx.x] += rs[threadIdx.x + st]; rq[threadIdx.x] += rq[threadIdx.x + st]; }
    __syncthreads();
  }
  if (threadIdx.x == 0) {
    const float inv = 1.f/(24.f*HW_);
    float mean = rs[0]*inv;
    stats[bg*2 + 0] = mean;
    stats[bg*2 + 1] = rq[0]*inv - mean*mean;
  }
}

// ---------------------------------------------------------------------------
// Fused GN + silu + residual + scale/shift + leaky-relu + softmax blend.
// Output in (B,C,H,W) order.
// ---------------------------------------------------------------------------
__global__ void fuse_kernel(const float* __restrict__ mspa, const float* __restrict__ mspe,
                            const float* __restrict__ x_spa, const float* __restrict__ x_spe,
                            const float* __restrict__ st_a, const float* __restrict__ st_e,
                            const float* __restrict__ ga, const float* __restrict__ ba,
                            const float* __restrict__ ge, const float* __restrict__ be,
                            const float* __restrict__ att,
                            const float* __restrict__ cwa, const float* __restrict__ cba,
                            const float* __restrict__ cwe, const float* __restrict__ cbe,
                            float* __restrict__ out) {
  size_t idx = (size_t)blockIdx.x*blockDim.x + threadIdx.x;
  if (idx >= (size_t)B_*C_*HW_) return;
  int hw = idx & 4095; size_t t = idx >> 12;
  int c = (int)(t % C_), b = (int)(t / C_);
  int g = c / 24;
  size_t mi = ((size_t)b*HW_ + hw)*C_ + c;

  float va = mspa[mi];
  float na = (va - st_a[(b*4 + g)*2]) * rsqrtf(st_a[(b*4 + g)*2 + 1] + 1e-5f) * ga[c] + ba[c];
  float sa = silu_(na) + x_spa[idx];

  float ve = mspe[mi];
  float ne = (ve - st_e[(b*4 + g)*2]) * rsqrtf(st_e[(b*4 + g)*2 + 1] + 1e-5f) * ge[c] + be[c];
  float se = x_spe[idx] + silu_(ne);

  float a0 = att[0], a1 = att[1];
  float mx = fmaxf(a0, a1);
  float e0 = __expf(a0 - mx), e1 = __expf(a1 - mx);
  float inv = 1.f/(e0 + e1);

  float fa = fmaf(sa, cwa[c], cba[c]); fa = fa > 0.f ? fa : 0.01f*fa;
  float fe = fmaf(se, cwe[c], cbe[c]); fe = fe > 0.f ? fe : 0.01f*fe;
  out[idx] = fa*(e0*inv) + fe*(e1*inv);
}

// ---------------------------------------------------------------------------
extern "C" void kernel_launch(void* const* d_in, const int* in_sizes, int n_in,
                              void* d_out, int out_size, void* d_ws, size_t ws_size,
                              hipStream_t stream) {
  const float* x_spa      = (const float*)d_in[1];
  const float* x_spe      = (const float*)d_in[2];
  const float* spa_in_w   = (const float*)d_in[3];
  const float* spa_conv_w = (const float*)d_in[4];
  const float* spa_conv_b = (const float*)d_in[5];
  const float* spa_xproj  = (const float*)d_in[6];
  const float* spa_dt_w   = (const float*)d_in[7];
  const float* spa_dt_b   = (const float*)d_in[8];
  const float* spa_Alog   = (const float*)d_in[9];
  const float* spa_D      = (const float*)d_in[10];
  const float* spa_out_w  = (const float*)d_in[11];
  const float* spe_in_w   = (const float*)d_in[12];
  const float* spe_conv_w = (const float*)d_in[13];
  const float* spe_conv_b = (const float*)d_in[14];
  const float* spe_xproj  = (const float*)d_in[15];
  const float* spe_dt_w   = (const float*)d_in[16];
  const float* spe_dt_b   = (const float*)d_in[17];
  const float* spe_Alog   = (const float*)d_in[18];
  const float* spe_D      = (const float*)d_in[19];
  const float* spe_out_w  = (const float*)d_in[20];
  const float* mlp_w1     = (const float*)d_in[21];
  const float* mlp_b1     = (const float*)d_in[22];
  const float* mlp_w2     = (const float*)d_in[23];
  const float* mlp_b2     = (const float*)d_in[24];
  const float* spa_gn_g   = (const float*)d_in[25];
  const float* spa_gn_b   = (const float*)d_in[26];
  const float* spe_gn_g   = (const float*)d_in[27];
  const float* spe_gn_b   = (const float*)d_in[28];
  const float* att_w      = (const float*)d_in[29];
  const float* cspa_w     = (const float*)d_in[30];
  const float* cspa_b     = (const float*)d_in[31];
  const float* cspe_w     = (const float*)d_in[32];
  const float* cspe_b     = (const float*)d_in[33];

  // ---- workspace carve-out (256B aligned regions) ----
  char* ws = (char*)d_ws;
  size_t off = 0;
  auto alloc = [&](size_t bytes) -> void* {
    void* ptr = ws + off;
    off = (off + bytes + 255) & ~(size_t)255;
    return ptr;
  };
  const size_t NTOK = (size_t)B_*HW_;
  float*    xc     = (float*)   alloc(NTOK*DIN_*4);
  _Float16* zh     = (_Float16*)alloc(NTOK*DIN_*2);
  float*    xcs    = (float*)   alloc(NTOK*DIN_*4);
  _Float16* xcsh   = (_Float16*)alloc(NTOK*DIN_*2);
  float*    dtb    = (float*)   alloc(NTOK*DIN_*4);
  float*    Bm     = (float*)   alloc(NTOK*NST_*4);
  float*    Cm     = (float*)   alloc(NTOK*NST_*4);
  float*    ysb    = (float*)   alloc(NTOK*DIN_*4);
  float*    mspa   = (float*)   alloc(NTOK*C_*4);
  float*    mspe   = (float*)   alloc(NTOK*C_*4);
  float*    carryH = (float*)   alloc((size_t)B_*NCH_*DIN_*NST_*4);
  float*    carryP = (float*)   alloc((size_t)B_*NCH_*DIN_*NST_*4);
  float*    hin    = (float*)   alloc((size_t)B_*NCH_*DIN_*NST_*4);
  float*    st_a   = (float*)   alloc(B_*4*2*4);
  float*    st_e   = (float*)   alloc(B_*4*2*4);
  _Float16* W1T    = (_Float16*)alloc(192*96*2);
  _Float16* W2T    = (_Float16*)alloc(96*192*2);
  _Float16* WinT   = (_Float16*)alloc(384*96*2);
  _Float16* WxT    = (_Float16*)alloc(48*192*2);
  _Float16* WoutT  = (_Float16*)alloc(96*192*2);
  (void)ws_size; (void)in_sizes; (void)n_in; (void)out_size;

  // 1) weight packing (f32 -> transposed f16)
  pack_weights_kernel<<<(96*384 + 255)/256, 256, 0, stream>>>(
      mlp_w1, mlp_w2, spa_in_w, spa_xproj, spa_out_w, W1T, W2T, WinT, WxT, WoutT);

  // 2) spatial MLP + in-projection  (2048 tiles, 4 waves/block)
  spa_mlp_inproj_kernel<<<512, 128, 0, stream>>>(
      x_spa, W1T, mlp_b1, W2T, mlp_b2, WinT, xc, zh);

  // 3) depthwise causal conv + silu
  spa_conv_kernel<<<(int)((NTOK*DIN_ + 255)/256), 256, 0, stream>>>(
      xc, spa_conv_w, spa_conv_b, xcs, xcsh);

  // 4) x-projection + dt
  spa_xproj_kernel<<<512, 128, 0, stream>>>(
      xcsh, WxT, spa_dt_w, spa_dt_b, dtb, Bm, Cm);

  // 5) chunked selective scan
  spa_scan_phase1<<<B_*NCH_, DIN_, 0, stream>>>(dtb, xcs, Bm, spa_Alog, carryH, carryP);
  spa_scan_phase2<<<B_, DIN_, 0, stream>>>(carryH, carryP, hin);
  spa_scan_phase3<<<B_*NCH_, DIN_, 0, stream>>>(dtb, xcs, Bm, Cm, spa_Alog, spa_D, hin, ysb);

  // 6) gating + out-projection
  spa_out_kernel<<<512, 128, 0, stream>>>(ysb, zh, WoutT, mspa);

  // 7) spectral mamba (independent of 2-6 except stream order; fills mspe)
  spe_mamba_kernel<<<4096, 256, 0, stream>>>(
      x_spe, spe_in_w, spe_conv_w, spe_conv_b, spe_xproj,
      spe_dt_w, spe_dt_b, spe_Alog, spe_D, spe_out_w, mspe);

  // 8) group-norm statistics
  gn_stats_kernel<<<B_*4, 256, 0, stream>>>(mspa, st_a);
  gn_stats_kernel<<<B_*4, 256, 0, stream>>>(mspe, st_e);

  // 9) fused normalization / residual / attention blend
  fuse_kernel<<<(int)(((size_t)B_*C_*HW_ + 255)/256), 256, 0, stream>>>(
      mspa, mspe, x_spa, x_spe, st_a, st_e,
      spa_gn_g, spa_gn_b, spe_gn_g, spe_gn_b,
      att_w, cspa_w, cspa_b, cspe_w, cspe_b, (float*)d_out);
}
